// IDAdapterPostfuseModule_45810121179378
// MI455X (gfx1250) — matrix operand order repaired and checked by reading, not compiled, not run on hardware
//
#include <hip/hip_runtime.h>
#include <hip/hip_bf16.h>
#include <math.h>

// ---------------- problem constants (from reference) ----------------
static constexpr int Bc = 64;
static constexpr int Sc = 1024;
static constexpr int Dc = 1024;
static constexpr int Mc = 8;
static constexpr int Tc = 8;
static constexpr int NF = Bc * Mc * Tc;   // 4096 fused tokens
static constexpr int BS = Bc * Sc;        // 65536
static constexpr int BM = Bc * Mc;        // 512

// ---------------- CDNA5 WMMA types ----------------
typedef __attribute__((ext_vector_type(16))) __bf16 v16bf;
typedef __attribute__((ext_vector_type(8)))  float  v8f;

union Frag16 {           // 16 bf16 per lane = 32 bytes = 8 VGPRs
    uint4 q[2];
    v16bf v;
};

// round-to-nearest-even f32 -> bf16 bits
__device__ __forceinline__ unsigned short f2bf(float f) {
    unsigned int u = __float_as_uint(f);
    u = (u + 0x7FFFu + ((u >> 16) & 1u)) >> 16;
    return (unsigned short)u;
}

__device__ __forceinline__ float gelu_exact(float x) {
    return 0.5f * x * (1.0f + erff(x * 0.70710678118654752440f));
}

// A fragment: lane-local K chunks {0..7} and {16..23} (shifted by half*8)
__device__ __forceinline__ void loadA(Frag16& f, const unsigned short* p) {
    f.q[0] = *reinterpret_cast<const uint4*>(p);
    f.q[1] = *reinterpret_cast<const uint4*>(p + 16);
}
// B fragment: this lane's N column, K rows {0..7} and {16..23} (shifted by half*8)
__device__ __forceinline__ void loadB(Frag16& f, const unsigned short* p) {
    unsigned short t[16];
    #pragma unroll
    for (int s = 0; s < 16; ++s) t[s] = p[(size_t)(s < 8 ? s : s + 8) * Dc];
    __builtin_memcpy(&f, t, 32);
}

__device__ __forceinline__ v8f wmma_bf16(const Frag16& a, const Frag16& b, v8f c) {
    return __builtin_amdgcn_wmma_f32_16x16x32_bf16(false, a.v, false, b.v,
                                                   (short)0, c, false, false);
}

// block-wide 2-value sum reduction (blockDim.x == 256)
__device__ __forceinline__ void block_reduce2(float& a, float& b) {
    __shared__ float sa[256];
    __shared__ float sb[256];
    int tid = threadIdx.x;
    sa[tid] = a; sb[tid] = b;
    __syncthreads();
    for (int off = 128; off > 0; off >>= 1) {
        if (tid < off) { sa[tid] += sa[tid + off]; sb[tid] += sb[tid + off]; }
        __syncthreads();
    }
    a = sa[0]; b = sb[0];
    __syncthreads();
}

// ---------------- 1) index construction (wave32 ballot scan) ----------------
__global__ void build_indices_kernel(const unsigned char* __restrict__ mask,
                                     const int* __restrict__ num_objects,
                                     int* __restrict__ pos,   // [NF]
                                     int* __restrict__ ord)   // [NF/T] valid (b,m) in order
{
    int lane = threadIdx.x;   // blockDim == 32
    int base = 0;
    for (int c = 0; c < BS; c += 32) {
        int idx = c + lane;
        bool m = mask[idx] != 0;
        unsigned int bal = (unsigned int)__ballot((int)m);
        int pre = __popc(bal & ((1u << lane) - 1u));
        if (m && (base + pre) < NF) pos[base + pre] = idx;
        base += __popc(bal);
    }
    base = 0;
    for (int c = 0; c < BM; c += 32) {
        int idx = c + lane;
        bool v = (idx % Mc) < num_objects[idx / Mc];
        unsigned int bal = (unsigned int)__ballot((int)v);
        int pre = __popc(bal & ((1u << lane) - 1u));
        if (v && (base + pre) < (NF / Tc)) ord[base + pre] = idx;
        base += __popc(bal);
    }
}

// ---------------- 2) f32 -> bf16 weight conversion ----------------
__global__ void f32_to_bf16_kernel(const float* __restrict__ src,
                                   unsigned short* __restrict__ dst, int n) {
    int i = blockIdx.x * blockDim.x + threadIdx.x;
    if (i < n) dst[i] = f2bf(src[i]);
}

// ---------------- 3) full passthrough copy of text_embeds ----------------
__global__ void copy_vec4_kernel(const float4* __restrict__ src,
                                 float4* __restrict__ dst, int n4) {
    int i = blockIdx.x * blockDim.x + threadIdx.x;
    int stride = gridDim.x * blockDim.x;
    for (; i < n4; i += stride) dst[i] = src[i];
}

// ---------------- 4) gather + concat + LN(2048) -> bf16 ----------------
__global__ __launch_bounds__(256)
void fuse_ln1_kernel(const float* __restrict__ text,     // [B*S, D]
                     const float* __restrict__ obj,      // [B*M*T, D]
                     const int* __restrict__ pos,
                     const int* __restrict__ ord,
                     const float* __restrict__ g,        // [2D]
                     const float* __restrict__ b,        // [2D]
                     unsigned short* __restrict__ x0n)   // [NF, 2D] bf16
{
    int i = blockIdx.x;
    const float* trow = text + (size_t)pos[i] * Dc;
    const float* orow = obj + ((size_t)ord[i >> 3] * Tc + (i & 7)) * Dc;

    float xv[8];
    float s = 0.f, s2 = 0.f;
    #pragma unroll
    for (int u = 0; u < 8; ++u) {
        int j = threadIdx.x + u * 256;
        float v = (j < Dc) ? trow[j] : orow[j - Dc];
        xv[u] = v; s += v; s2 += v * v;
    }
    block_reduce2(s, s2);
    float mu = s * (1.0f / (2 * Dc));
    float var = s2 * (1.0f / (2 * Dc)) - mu * mu;
    float rs = rsqrtf(var + 1e-5f);
    #pragma unroll
    for (int u = 0; u < 8; ++u) {
        int j = threadIdx.x + u * 256;
        float nv = (xv[u] - mu) * rs * g[j] + b[j];
        x0n[(size_t)i * (2 * Dc) + j] = f2bf(nv);
    }
}

// ---------------- 5) WMMA GEMM: C = act(A[N,K] @ W[K,1024] + bias) ----------------
// One 16x64 output tile per wave: 4 accumulators share one A fragment.
// K unrolled by 2 with ping-pong register buffers (no copy-rotation).
// Requires K % 64 == 0. mode 1: GELU -> bf16 out; mode 0: f32 out.
__global__ __launch_bounds__(256)
void gemm_wmma_kernel(const unsigned short* __restrict__ A,   // bf16 [N, K]
                      const unsigned short* __restrict__ W,   // bf16 [K, 1024]
                      const float* __restrict__ bias,         // [1024]
                      float* __restrict__ outF,
                      unsigned short* __restrict__ outB,
                      int K, int mode)
{
    const int Dout = Dc;            // 1024
    const int tilesX = Dout / 64;   // 16 column-groups of 64
    int lane = threadIdx.x & 31;
    int wave = threadIdx.x >> 5;
    int tile = blockIdx.x * 8 + wave;
    int ty = tile / tilesX;         // 16-row tile
    int tx = tile % tilesX;         // 64-col group

    int half = lane >> 4;    // 0 or 1
    int l16  = lane & 15;

    // A layout (16-bit A 16x32): lane<16 -> row l16, K {0..7,16..23};
    // lane>=16 -> row l16, K {8..15,24..31}.
    const unsigned short* arow = A + (size_t)(ty * 16 + l16) * K + half * 8;
    // B layout (16-bit B 32x16): lane holds column l16, K rows {0..7,16..23} (+half*8).
    const unsigned short* wbase = W + (size_t)(half * 8) * Dout + tx * 64 + l16;

    v8f acc0 = {0.f, 0.f, 0.f, 0.f, 0.f, 0.f, 0.f, 0.f};
    v8f acc1 = acc0, acc2 = acc0, acc3 = acc0;

    Frag16 a0, b00, b01, b02, b03;   // buffer 0
    Frag16 a1, b10, b11, b12, b13;   // buffer 1

    // preload K-step 0 into buffer 0
    loadA(a0, arow);
    loadB(b00, wbase +  0);
    loadB(b01, wbase + 16);
    loadB(b02, wbase + 32);
    loadB(b03, wbase + 48);

    for (int k0 = 0; k0 < K; k0 += 64) {
        // issue loads for K-step k0+32 into buffer 1
        {
            const unsigned short* ap = arow + (k0 + 32);
            const unsigned short* wp = wbase + (size_t)(k0 + 32) * Dout;
            loadA(a1, ap);
            loadB(b10, wp +  0);
            loadB(b11, wp + 16);
            loadB(b12, wp + 32);
            loadB(b13, wp + 48);
        }
        // compute on buffer 0
        acc0 = wmma_bf16(a0, b00, acc0);
        acc1 = wmma_bf16(a0, b01, acc1);
        acc2 = wmma_bf16(a0, b02, acc2);
        acc3 = wmma_bf16(a0, b03, acc3);
        // issue loads for K-step k0+64 into buffer 0 (guarded)
        if (k0 + 64 < K) {
            const unsigned short* ap = arow + (k0 + 64);
            const unsigned short* wp = wbase + (size_t)(k0 + 64) * Dout;
            loadA(a0, ap);
            loadB(b00, wp +  0);
            loadB(b01, wp + 16);
            loadB(b02, wp + 32);
            loadB(b03, wp + 48);
        }
        // compute on buffer 1
        acc0 = wmma_bf16(a1, b10, acc0);
        acc1 = wmma_bf16(a1, b11, acc1);
        acc2 = wmma_bf16(a1, b12, acc2);
        acc3 = wmma_bf16(a1, b13, acc3);
    }

    // C/D layout: VGPR r -> M=r (lanes 0-15, N=lane) / M=r+8 (lanes 16-31, N=lane-16)
    int rowb = ty * 16 + half * 8;
    int colb = tx * 64 + l16;
    auto epilogue = [&](v8f acc, int n) {
        float bcol = bias[n];
        #pragma unroll
        for (int r = 0; r < 8; ++r) {
            size_t o = (size_t)(rowb + r) * Dout + n;
            float v = acc[r] + bcol;
            if (mode == 1) outB[o] = f2bf(gelu_exact(v));
            else           outF[o] = v;
        }
    };
    epilogue(acc0, colb);
    epilogue(acc1, colb + 16);
    epilogue(acc2, colb + 32);
    epilogue(acc3, colb + 48);
}

// ---------------- 6) x1 = y1 + t ; LN(ln2) -> bf16 ----------------
__global__ __launch_bounds__(256)
void res_ln2_kernel(const float* __restrict__ yf,      // [NF, D]
                    const float* __restrict__ text,    // [B*S, D]
                    const int* __restrict__ pos,
                    const float* __restrict__ g,
                    const float* __restrict__ b,
                    float* __restrict__ x1,            // [NF, D] f32
                    unsigned short* __restrict__ x1n)  // [NF, D] bf16
{
    int i = blockIdx.x;
    const float* trow = text + (size_t)pos[i] * Dc;
    float xv[4];
    float s = 0.f, s2 = 0.f;
    #pragma unroll
    for (int u = 0; u < 4; ++u) {
        int j = threadIdx.x + u * 256;
        float v = yf[(size_t)i * Dc + j] + trow[j];
        x1[(size_t)i * Dc + j] = v;
        xv[u] = v; s += v; s2 += v * v;
    }
    block_reduce2(s, s2);
    float mu = s * (1.0f / Dc);
    float var = s2 * (1.0f / Dc) - mu * mu;
    float rs = rsqrtf(var + 1e-5f);
    #pragma unroll
    for (int u = 0; u < 4; ++u) {
        int j = threadIdx.x + u * 256;
        float nv = (xv[u] - mu) * rs * g[j] + b[j];
        x1n[(size_t)i * Dc + j] = f2bf(nv);
    }
}

// ---------------- 7) x2 = y2 + x1 ; LN(lnf) ; scatter to out[pos] ----------------
__global__ __launch_bounds__(256)
void final_ln_scatter_kernel(const float* __restrict__ yf,   // [NF, D]
                             const float* __restrict__ x1,   // [NF, D]
                             const int* __restrict__ pos,
                             const float* __restrict__ g,
                             const float* __restrict__ b,
                             float* __restrict__ out)        // [B*S, D]
{
    int i = blockIdx.x;
    float xv[4];
    float s = 0.f, s2 = 0.f;
    #pragma unroll
    for (int u = 0; u < 4; ++u) {
        int j = threadIdx.x + u * 256;
        float v = yf[(size_t)i * Dc + j] + x1[(size_t)i * Dc + j];
        xv[u] = v; s += v; s2 += v * v;
    }
    block_reduce2(s, s2);
    float mu = s * (1.0f / Dc);
    float var = s2 * (1.0f / Dc) - mu * mu;
    float rs = rsqrtf(var + 1e-5f);
    float* orow = out + (size_t)pos[i] * Dc;
    #pragma unroll
    for (int u = 0; u < 4; ++u) {
        int j = threadIdx.x + u * 256;
        orow[j] = (xv[u] - mu) * rs * g[j] + b[j];
    }
}

// ---------------- host-side launch ----------------
extern "C" void kernel_launch(void* const* d_in, const int* in_sizes, int n_in,
                              void* d_out, int out_size, void* d_ws, size_t ws_size,
                              hipStream_t stream) {
    const float* text  = (const float*)d_in[0];   // [B,S,D]
    const float* obj   = (const float*)d_in[1];   // [B,M,T,D]
    const float* ln1_g = (const float*)d_in[2];
    const float* ln1_b = (const float*)d_in[3];
    const float* w1a   = (const float*)d_in[4];   // [2D, D]
    const float* b1a   = (const float*)d_in[5];
    const float* w1b   = (const float*)d_in[6];   // [D, D]
    const float* b1b   = (const float*)d_in[7];
    const float* ln2_g = (const float*)d_in[8];
    const float* ln2_b = (const float*)d_in[9];
    const float* w2a   = (const float*)d_in[10];
    const float* b2a   = (const float*)d_in[11];
    const float* w2b   = (const float*)d_in[12];
    const float* b2b   = (const float*)d_in[13];
    const float* lnf_g = (const float*)d_in[14];
    const float* lnf_b = (const float*)d_in[15];
    const unsigned char* mask = (const unsigned char*)d_in[16];  // bool [B,S]
    const int* num_objects    = (const int*)d_in[17];            // [B]
    float* out = (float*)d_out;

    // workspace carve-out (256B aligned)
    char* ws = (char*)d_ws;
    size_t off = 0;
    auto take = [&](size_t bytes) -> void* {
        void* p = ws + off;
        off = (off + bytes + 255) & ~(size_t)255;
        return p;
    };
    int*            pos  = (int*)take(NF * sizeof(int));
    int*            ord  = (int*)take((NF / Tc) * sizeof(int));
    unsigned short* W1a  = (unsigned short*)take((size_t)2 * Dc * Dc * 2);
    unsigned short* W1b  = (unsigned short*)take((size_t)Dc * Dc * 2);
    unsigned short* W2a  = (unsigned short*)take((size_t)Dc * Dc * 2);
    unsigned short* W2b  = (unsigned short*)take((size_t)Dc * Dc * 2);
    unsigned short* x0n  = (unsigned short*)take((size_t)NF * 2 * Dc * 2);
    unsigned short* hbuf = (unsigned short*)take((size_t)NF * Dc * 2);
    float*          yf   = (float*)take((size_t)NF * Dc * 4);
    float*          x1   = (float*)take((size_t)NF * Dc * 4);
    unsigned short* x1n  = (unsigned short*)take((size_t)NF * Dc * 2);
    (void)ws_size; (void)n_in; (void)in_sizes; (void)out_size;

    // 1) indices
    build_indices_kernel<<<1, 32, 0, stream>>>(mask, num_objects, pos, ord);

    // 2) weight conversion
    {
        int n1 = 2 * Dc * Dc, n2 = Dc * Dc;
        f32_to_bf16_kernel<<<(n1 + 255) / 256, 256, 0, stream>>>(w1a, W1a, n1);
        f32_to_bf16_kernel<<<(n2 + 255) / 256, 256, 0, stream>>>(w1b, W1b, n2);
        f32_to_bf16_kernel<<<(n2 + 255) / 256, 256, 0, stream>>>(w2a, W2a, n2);
        f32_to_bf16_kernel<<<(n2 + 255) / 256, 256, 0, stream>>>(w2b, W2b, n2);
    }

    // 3) passthrough copy (bandwidth-bound part)
    {
        int n4 = (BS * Dc) / 4;
        copy_vec4_kernel<<<8192, 256, 0, stream>>>(
            (const float4*)text, (float4*)out, n4);
    }

    // 4) gather + concat + LN1 (bf16 activations)
    fuse_ln1_kernel<<<NF, 256, 0, stream>>>(text, obj, pos, ord, ln1_g, ln1_b, x0n);

    const int gemm_blocks = (NF / 16) * (Dc / 64) / 8;   // 512

    // 5) h1 = gelu(x0n @ W1a + b1a)
    gemm_wmma_kernel<<<gemm_blocks, 256, 0, stream>>>(x0n, W1a, b1a, nullptr, hbuf, 2 * Dc, 1);
    // 6) y1 = h1 @ W1b + b1b
    gemm_wmma_kernel<<<gemm_blocks, 256, 0, stream>>>(hbuf, W1b, b1b, yf, nullptr, Dc, 0);
    // 7) x1 = y1 + t ; LN2 -> x1n
    res_ln2_kernel<<<NF, 256, 0, stream>>>(yf, text, pos, ln2_g, ln2_b, x1, x1n);
    // 8) h2 = gelu(x1n @ W2a + b2a)
    gemm_wmma_kernel<<<gemm_blocks, 256, 0, stream>>>(x1n, W2a, b2a, nullptr, hbuf, Dc, 1);
    // 9) y2 = h2 @ W2b + b2b
    gemm_wmma_kernel<<<gemm_blocks, 256, 0, stream>>>(hbuf, W2b, b2b, yf, nullptr, Dc, 0);
    // 10) x2 = y2 + x1 ; final LN ; scatter into out
    final_ln_scatter_kernel<<<NF, 256, 0, stream>>>(yf, x1, pos, lnf_g, lnf_b, out);
}
